// EnvironmentConsistentAttention_13434657702214
// MI455X (gfx1250) — matrix operand order repaired and checked by reading, not compiled, not run on hardware
//
#include <hip/hip_runtime.h>
#include <cstdint>
#include <cstddef>

// Problem constants (from reference): B=4, H=W=32, C=256
#define BH 32
#define BW 32
#define BC 256
#define NB 4
#define FF 2304   // 9*C patch feature dim
#define LL 992    // Hp*Wp for both crop directions (31*32 = 32*31)
#define EPSN 1e-4f
#define SMS 10.0f

typedef __attribute__((ext_vector_type(16))) __bf16 v16bf;
typedef __attribute__((ext_vector_type(8)))  float  v8f;

union FragB { v16bf v; unsigned int u[8]; };

__device__ __forceinline__ unsigned short f2bf(float x) {
    union { float f; unsigned u; } v; v.f = x;
    unsigned r = v.u + 0x7FFFu + ((v.u >> 16) & 1u);   // round-to-nearest-even
    return (unsigned short)(r >> 16);
}

// A-matrix 16x32 bf16 fragment (per CDNA5 ISA lane layout), pair-contiguous rows
__device__ __forceinline__ void loadA(FragB& f, const unsigned short* row, int k, int half) {
#pragma unroll
    for (int v = 0; v < 8; ++v) {
        int ka = ((v < 4) ? 2 * v : 16 + 2 * (v - 4)) + half * 8;
        f.u[v] = *(const unsigned int*)(row + k + ka);
    }
}
// B-matrix 32x16 bf16 fragment, pair-contiguous along K in the source row
__device__ __forceinline__ void loadBf(FragB& f, const unsigned short* row, int k, int half) {
#pragma unroll
    for (int v = 0; v < 8; ++v) {
        int kb = ((v < 4) ? 2 * v : 16 + 2 * (v - 4)) + half * 8;
        f.u[v] = *(const unsigned int*)(row + k + kb);
    }
}

// crop-frame -> full image index.  dir 0: a=left(x[:-1]), b=right(x[1:])
//                                  dir 1: a=top (x[:,1:]), b=bottom(x[:,:-1])
__device__ __forceinline__ int img_index(int b, int dir, int which, int hc, int wc, int c) {
    int h, w;
    if (dir == 0) { h = hc + which;       w = wc; }
    else          { h = hc;               w = wc + (which ^ 1); }
    return ((b * BH + h) * BW + w) * BC + c;
}

// ---------------- per-pixel squared channel norms -------------------------
__global__ void eca_sqnorm(const float* __restrict__ x, float* __restrict__ sa,
                           float* __restrict__ sb, int b, int dir, int Hp, int Wp) {
    __shared__ float ra[256];
    __shared__ float rb[256];
    int l = blockIdx.x;               // 0..LL-1
    int hc = l / Wp, wc = l % Wp;
    int c = threadIdx.x;              // 256 channels
    float va = x[img_index(b, dir, 0, hc, wc, c)];
    float vb = x[img_index(b, dir, 1, hc, wc, c)];
    ra[c] = va * va; rb[c] = vb * vb;
    __syncthreads();
    for (int s = 128; s > 0; s >>= 1) {
        if (c < s) { ra[c] += ra[c + s]; rb[c] += rb[c + s]; }
        __syncthreads();
    }
    if (c == 0) { sa[l] = ra[0]; sb[l] = rb[0]; }
}

// ---------------- 3x3 box-sum -> inverse patch norms ----------------------
__global__ void eca_invnorm(const float* __restrict__ sa, const float* __restrict__ sb,
                            float* __restrict__ ia, float* __restrict__ ib,
                            int Hp, int Wp) {
    int l = blockIdx.x * blockDim.x + threadIdx.x;
    if (l >= Hp * Wp) return;
    int hc = l / Wp, wc = l % Wp;
    float suma = 0.f, sumb = 0.f;
    for (int p = 0; p < 3; ++p) {
        int hh = hc + p - 1;
        if (hh < 0 || hh >= Hp) continue;
        for (int q = 0; q < 3; ++q) {
            int ww = wc + q - 1;
            if (ww < 0 || ww >= Wp) continue;
            suma += sa[hh * Wp + ww];
            sumb += sb[hh * Wp + ww];
        }
    }
    ia[l] = 1.0f / fmaxf(sqrtf(suma), EPSN);
    ib[l] = 1.0f / fmaxf(sqrtf(sumb), EPSN);
}

// ---------------- build Pa, Pb (F x L bf16) and yT (L x F bf16) -----------
__global__ void eca_build(const float* __restrict__ x,
                          unsigned short* __restrict__ Pa, unsigned short* __restrict__ Pb,
                          unsigned short* __restrict__ yT,
                          const float* __restrict__ ia, const float* __restrict__ ib,
                          int b, int dir, int Hp, int Wp) {
    int t = blockIdx.x * 256 + threadIdx.x;     // exactly FF*LL threads
    int l = t % LL;
    int f = t / LL;
    int p = f / (3 * BC);
    int r = f % (3 * BC);
    int q = r / BC;
    int c = r % BC;
    int hc = l / Wp, wc = l % Wp;
    int hh = hc + p - 1, ww = wc + q - 1;
    float va = 0.f, vb = 0.f;
    if (hh >= 0 && hh < Hp && ww >= 0 && ww < Wp) {
        va = x[img_index(b, dir, 0, hh, ww, c)];
        vb = x[img_index(b, dir, 1, hh, ww, c)];
    }
    Pa[(size_t)f * LL + l] = f2bf(va);
    Pb[(size_t)f * LL + l] = f2bf(vb);
    yT[(size_t)l * FF + f] = f2bf(va * vb * ia[l] * ib[l]);
}

// ---------------- GEMM1: logits(L x L) = yT * yT^T  (K = F) ---------------
// 2x2 register blocking: each wave computes a 32x32 output block.
// 8 x b128 fragment loads feed 4 WMMAs -> 2 loads/WMMA.
__global__ void eca_gemm1(const unsigned short* __restrict__ yT,
                          float* __restrict__ logits) {
    const int NTM = (LL / 16) / 2;               // 31 macro tiles each way
    int tile = blockIdx.x * 8 + (threadIdx.x >> 5);
    if (tile >= NTM * NTM) return;               // wave-uniform
    int tmm = tile / NTM, tnn = tile % NTM;
    int ln = threadIdx.x & 31;
    int q = ln & 15, half = ln >> 4;
    int m0 = tmm * 32, n0 = tnn * 32;
    const unsigned short* A0 = yT + (size_t)(m0 + q) * FF;
    const unsigned short* A1 = yT + (size_t)(m0 + 16 + q) * FF;
    const unsigned short* B0 = yT + (size_t)(n0 + q) * FF;
    const unsigned short* B1 = yT + (size_t)(n0 + 16 + q) * FF;
    v8f c00 = {}, c01 = {}, c10 = {}, c11 = {};
    for (int k = 0; k < FF; k += 32) {
        __builtin_prefetch(A0 + k + 256, 0, 3);
        __builtin_prefetch(B0 + k + 256, 0, 3);
        FragB a0, a1, b0, b1;
        loadA(a0, A0, k, half);
        loadA(a1, A1, k, half);
        loadBf(b0, B0, k, half);
        loadBf(b1, B1, k, half);
        c00 = __builtin_amdgcn_wmma_f32_16x16x32_bf16(false, a0.v, false, b0.v, (short)0, c00, false, false);
        c01 = __builtin_amdgcn_wmma_f32_16x16x32_bf16(false, a0.v, false, b1.v, (short)0, c01, false, false);
        c10 = __builtin_amdgcn_wmma_f32_16x16x32_bf16(false, a1.v, false, b0.v, (short)0, c10, false, false);
        c11 = __builtin_amdgcn_wmma_f32_16x16x32_bf16(false, a1.v, false, b1.v, (short)0, c11, false, false);
    }
#pragma unroll
    for (int r = 0; r < 8; ++r) {
        size_t row0 = (size_t)(m0 + r + 8 * half) * LL;
        size_t row1 = (size_t)(m0 + 16 + r + 8 * half) * LL;
        logits[row0 + n0 + q]      = c00[r];
        logits[row0 + n0 + 16 + q] = c01[r];
        logits[row1 + n0 + q]      = c10[r];
        logits[row1 + n0 + 16 + q] = c11[r];
    }
}

// ---------------- row softmax (scale 10) -> bf16 --------------------------
__global__ void eca_softmax(const float* __restrict__ logits,
                            unsigned short* __restrict__ sm) {
    __shared__ float red[256];
    int row = blockIdx.x;
    int tid = threadIdx.x;
    const float* src = logits + (size_t)row * LL;
    float m = -1e30f;
    for (int l = tid; l < LL; l += 256) m = fmaxf(m, src[l]);
    red[tid] = m; __syncthreads();
    for (int s = 128; s > 0; s >>= 1) {
        if (tid < s) red[tid] = fmaxf(red[tid], red[tid + s]);
        __syncthreads();
    }
    float rowm = red[0];
    __syncthreads();
    float sum = 0.f;
    for (int l = tid; l < LL; l += 256) sum += __expf((src[l] - rowm) * SMS);
    red[tid] = sum; __syncthreads();
    for (int s = 128; s > 0; s >>= 1) {
        if (tid < s) red[tid] += red[tid + s];
        __syncthreads();
    }
    float inv = 1.0f / red[0];
    unsigned short* dst = sm + (size_t)row * LL;
    for (int l = tid; l < LL; l += 256)
        dst[l] = f2bf(__expf((src[l] - rowm) * SMS) * inv);
}

// ---------------- GEMM2: Za = Pa * att^T, Zb = Pb * att^T  (K = L) --------
// Each wave: 1 M-tile x 2 N-tiles for BOTH matrices -> B fragments shared,
// 8 x b128 loads feed 4 WMMAs.
__global__ void eca_gemm2(const unsigned short* __restrict__ Pa,
                          const unsigned short* __restrict__ Pb,
                          const unsigned short* __restrict__ sm,
                          float* __restrict__ Za, float* __restrict__ Zb) {
    const int MT = FF / 16;                      // 144
    const int NTN = (LL / 16) / 2;               // 31 macro N tiles
    int tile = blockIdx.x * 8 + (threadIdx.x >> 5);
    if (tile >= MT * NTN) return;                // wave-uniform
    int tm = tile / NTN, tnn = tile % NTN;
    int ln = threadIdx.x & 31;
    int q = ln & 15, half = ln >> 4;
    int m = tm * 16 + q;
    int n0 = tnn * 32;
    const unsigned short* Aa = Pa + (size_t)m * LL;
    const unsigned short* Ab = Pb + (size_t)m * LL;
    const unsigned short* B0 = sm + (size_t)(n0 + q) * LL;
    const unsigned short* B1 = sm + (size_t)(n0 + 16 + q) * LL;
    v8f ca0 = {}, ca1 = {}, cb0 = {}, cb1 = {};
    for (int k = 0; k < LL; k += 32) {
        __builtin_prefetch(Aa + k + 256, 0, 3);
        __builtin_prefetch(B0 + k + 256, 0, 3);
        FragB fa, fb, b0, b1;
        loadA(fa, Aa, k, half);
        loadA(fb, Ab, k, half);
        loadBf(b0, B0, k, half);
        loadBf(b1, B1, k, half);
        ca0 = __builtin_amdgcn_wmma_f32_16x16x32_bf16(false, fa.v, false, b0.v, (short)0, ca0, false, false);
        ca1 = __builtin_amdgcn_wmma_f32_16x16x32_bf16(false, fa.v, false, b1.v, (short)0, ca1, false, false);
        cb0 = __builtin_amdgcn_wmma_f32_16x16x32_bf16(false, fb.v, false, b0.v, (short)0, cb0, false, false);
        cb1 = __builtin_amdgcn_wmma_f32_16x16x32_bf16(false, fb.v, false, b1.v, (short)0, cb1, false, false);
    }
#pragma unroll
    for (int r = 0; r < 8; ++r) {
        size_t row = (size_t)(tm * 16 + r + 8 * half) * LL;
        Za[row + n0 + q]      = ca0[r];
        Za[row + n0 + 16 + q] = ca1[r];
        Zb[row + n0 + q]      = cb0[r];
        Zb[row + n0 + 16 + q] = cb1[r];
    }
}

// ---------------- fold (conv-transpose scatter) + stitch ------------------
__device__ __forceinline__ float foldZ(const float* __restrict__ Z,
                                       int hc, int wc, int c, int Hp, int Wp) {
    float s = 0.f;
    for (int p = 0; p < 3; ++p) {
        int hh = hc + 1 - p;
        if (hh < 0 || hh >= Hp) continue;
        for (int q = 0; q < 3; ++q) {
            int ww = wc + 1 - q;
            if (ww < 0 || ww >= Wp) continue;
            int f = (p * 3 + q) * BC + c;
            s += Z[(size_t)f * LL + hh * Wp + ww];
        }
    }
    return s;
}

__global__ void eca_fold(const float* __restrict__ Za, const float* __restrict__ Zb,
                         float* __restrict__ out, int b, int dir) {
    int t = blockIdx.x * 256 + threadIdx.x;      // 32*32*256 threads
    int c = t % BC;
    int rest = t / BC;
    int w = rest % BW;
    int h = rest / BW;
    size_t oidx = ((size_t)(b * BH + h) * BW + w) * BC + c;
    float v;
    if (dir == 0) {                 // left/right: Hp=31, Wp=32; ya=yl, yb=yr
        if (h == 0)        v = 0.5f  * foldZ(Zb, 0,     w, c, 31, 32);
        else if (h == 31)  v = 0.5f  * foldZ(Za, 30,    w, c, 31, 32);
        else               v = 0.25f * (foldZ(Zb, h,    w, c, 31, 32) +
                                        foldZ(Za, h - 1, w, c, 31, 32));
        out[oidx] = v;              // first pass: overwrite (covers all elems)
    } else {                        // top/bottom: Hp=32, Wp=31; ya=yt, yb=ybot
        if (w == 0)        v = 0.5f  * foldZ(Za, h, 0,     c, 32, 31);
        else if (w == 31)  v = 0.5f  * foldZ(Zb, h, 30,    c, 32, 31);
        else               v = 0.25f * (foldZ(Za, h, w,    c, 32, 31) +
                                        foldZ(Zb, h, w - 1, c, 32, 31));
        out[oidx] += v;             // second pass: accumulate (stream-ordered)
    }
}

extern "C" void kernel_launch(void* const* d_in, const int* in_sizes, int n_in,
                              void* d_out, int out_size, void* d_ws, size_t ws_size,
                              hipStream_t stream) {
    const float* x = (const float*)d_in[0];     // (4,32,32,256) f32; mask unused
    float* out = (float*)d_out;

    char* ws = (char*)d_ws;
    size_t off = 0;
    auto take = [&](size_t bytes) -> char* {
        char* p = ws + off;
        off = (off + bytes + 255) & ~(size_t)255;
        return p;
    };
    unsigned short* Pa  = (unsigned short*)take((size_t)FF * LL * 2);
    unsigned short* Pb  = (unsigned short*)take((size_t)FF * LL * 2);
    unsigned short* yT  = (unsigned short*)take((size_t)LL * FF * 2);
    float*          Log = (float*)take((size_t)LL * LL * 4);
    unsigned short* Sm  = (unsigned short*)take((size_t)LL * LL * 2);
    float*          Za  = (float*)take((size_t)FF * LL * 4);
    float*          Zb  = (float*)take((size_t)FF * LL * 4);
    float*          Sa  = (float*)take((size_t)LL * 4);
    float*          Sb  = (float*)take((size_t)LL * 4);
    float*          Ia  = (float*)take((size_t)LL * 4);
    float*          Ib  = (float*)take((size_t)LL * 4);

    const int NTM = (LL / 16) / 2;   // 31
    const int MT  = FF / 16;         // 144
    const int g_gemm1 = (NTM * NTM + 7) / 8;        // 121 blocks of 8 waves
    const int g_gemm2 = (MT * NTM + 7) / 8;         // 558 blocks (exact)
    const int g_build = (FF * LL) / 256;            // 8928 blocks (exact)
    const int g_fold  = (BH * BW * BC) / 256;       // 1024 blocks (exact)

    for (int b = 0; b < NB; ++b) {
        for (int dir = 0; dir < 2; ++dir) {
            int Hp = dir ? 32 : 31;
            int Wp = dir ? 31 : 32;
            eca_sqnorm <<<LL,      256, 0, stream>>>(x, Sa, Sb, b, dir, Hp, Wp);
            eca_invnorm<<<4,       256, 0, stream>>>(Sa, Sb, Ia, Ib, Hp, Wp);
            eca_build  <<<g_build, 256, 0, stream>>>(x, Pa, Pb, yT, Ia, Ib, b, dir, Hp, Wp);
            eca_gemm1  <<<g_gemm1, 256, 0, stream>>>(yT, Log);
            eca_softmax<<<LL,      256, 0, stream>>>(Log, Sm);
            eca_gemm2  <<<g_gemm2, 256, 0, stream>>>(Pa, Pb, Sm, Za, Zb);
            eca_fold   <<<g_fold,  256, 0, stream>>>(Za, Zb, out, b, dir);
        }
    }
}
